// Int8DynActInt4WeightLinear_3212635537641
// MI455X (gfx1250) — 8-bit matrix operand order repaired and checked by reading, compiled, not run on hardware
//
#include <hip/hip_runtime.h>

#define TOK_I 4096
#define GS    256
#define NG    16

#define LDA         272
#define ATILE_BYTES (128 * LDA)
#define BBASE       (2 * ATILE_BYTES)
#define LDS_TOTAL   (4 * ATILE_BYTES)

typedef __attribute__((ext_vector_type(8))) int   v8i;
typedef __attribute__((ext_vector_type(8))) float v8f;

__global__ __launch_bounds__(256) void quant_x_kernel(
    const float* __restrict__ x, signed char* __restrict__ qx,
    float* __restrict__ sx, float* __restrict__ zpx, int* __restrict__ rsx)
{
    const int t   = blockIdx.x;
    const int tid = threadIdx.x;
    const float* xr = x + (size_t)t * TOK_I;

    float v[NG];
    float mn = 0.0f, mx = 0.0f;
#pragma unroll
    for (int k = 0; k < NG; ++k) {
        float f = xr[tid + 256 * k];
        v[k] = f;
        mn = fminf(mn, f);
        mx = fmaxf(mx, f);
    }
    __shared__ float smn[256], smx[256];
    smn[tid] = mn; smx[tid] = mx;
    __syncthreads();
    for (int s = 128; s > 0; s >>= 1) {
        if (tid < s) {
            smn[tid] = fminf(smn[tid], smn[tid + s]);
            smx[tid] = fmaxf(smx[tid], smx[tid + s]);
        }
        __syncthreads();
    }
    __shared__ float s_scale, s_zp;
    __shared__ int   s_rs[NG];
    if (tid < NG) s_rs[tid] = 0;
    if (tid == 0) {
        float tmn = smn[0], tmx = smx[0];
        float scale = fmaxf((tmx - tmn) * (1.0f / 255.0f), 1.1920929e-7f);
        float dmin = tmn / scale, dmax = tmx / scale;
        float zp = ((-128.0f + dmin) + (127.0f + dmax) > 0.0f)
                       ? (-128.0f - dmin) : (127.0f - dmax);
        zp = rintf(fminf(fmaxf(zp, -128.0f), 127.0f));
        s_scale = scale; s_zp = zp;
    }
    __syncthreads();
    const float scale = s_scale, zp = s_zp;
#pragma unroll
    for (int k = 0; k < NG; ++k) {
        float qf = rintf(v[k] / scale) + zp;
        qf = fminf(fmaxf(qf, -128.0f), 127.0f);
        int q = (int)qf;
        qx[(size_t)t * TOK_I + tid + 256 * k] = (signed char)q;
        atomicAdd(&s_rs[k], q);
    }
    __syncthreads();
    if (tid < NG) rsx[t * NG + tid] = s_rs[tid];
    if (tid == 0) { sx[t] = scale; zpx[t] = zp; }
}

__global__ __launch_bounds__(256) void prep_w_kernel(
    const int* __restrict__ w, const float* __restrict__ scales,
    const float* __restrict__ zeros, signed char* __restrict__ qw,
    float2* __restrict__ wmeta, float* __restrict__ W0)
{
    const int o = blockIdx.x, tid = threadIdx.x;
    __shared__ int   cs[NG];
    __shared__ float sw0[NG];
    if (tid < NG) cs[tid] = 0;
    __syncthreads();
#pragma unroll
    for (int k = 0; k < NG; ++k) {
        int wv = w[(size_t)o * TOK_I + tid + 256 * k];
        qw[(size_t)o * TOK_I + tid + 256 * k] = (signed char)wv;
        atomicAdd(&cs[k], wv);
    }
    __syncthreads();
    if (tid < NG) {
        int g = tid;
        float sw = scales[o * NG + g];
        float zw = zeros[o * NG + g];
        wmeta[o * NG + g] = make_float2(sw, sw * zw);
        sw0[g] = sw * ((float)cs[g] - 256.0f * zw);
    }
    __syncthreads();
    if (tid == 0) {
        float acc = 0.0f;
#pragma unroll
        for (int g = 0; g < NG; ++g) acc += sw0[g];
        W0[o] = acc;
    }
}

__global__ __launch_bounds__(256) void gemm_kernel(
    const signed char* __restrict__ qx, const signed char* __restrict__ qw,
    const float* __restrict__ sx, const float* __restrict__ zpx,
    const int* __restrict__ rsx, const float2* __restrict__ wmeta,
    const float* __restrict__ W0, float* __restrict__ out, int O)
{
    extern __shared__ char lds[];

    const int I = TOK_I;
    const int tid  = threadIdx.x;
    const int wave = tid >> 5, lane = tid & 31;
    const int ln = lane & 15, hi = lane >> 4;
    const int mw = wave & 3, nw = wave >> 2;
    const int Tblk = blockIdx.x * 128, Oblk = blockIdx.y * 128;
    const int mbase = Tblk + mw * 32;
    const int nbase = Oblk + nw * 64;

    const signed char* asrc = qx + (size_t)Tblk * I;
    const signed char* bsrc = qw + (size_t)Oblk * I;

    const int ccol  = (tid & 15) << 4;
    const int crow0 = tid >> 4;

    int orow[4];
#pragma unroll
    for (int ni = 0; ni < 4; ++ni) orow[ni] = nbase + ni * 16 + ln;

    auto issue_stage = [&](int g, int buf) {
        const int k0 = g * GS;
#pragma unroll
        for (int it = 0; it < 8; ++it) {
            const int row = crow0 + 16 * it;
            const signed char* ga = asrc + (size_t)row * I + k0 + ccol;
            unsigned la = (unsigned)(buf * ATILE_BYTES + row * LDA + ccol);
            asm volatile("global_load_async_to_lds_b128 %0, %1, off"
                         :: "v"(la), "v"(ga) : "memory");
            const signed char* gb = bsrc + (size_t)row * I + k0 + ccol;
            unsigned lb = (unsigned)(BBASE + buf * ATILE_BYTES + row * LDA + ccol);
            asm volatile("global_load_async_to_lds_b128 %0, %1, off"
                         :: "v"(lb), "v"(gb) : "memory");
        }
        if (g + 1 < NG) {
            __builtin_prefetch(asrc + (size_t)crow0 * I + (g + 1) * GS + ccol, 0, 1);
            __builtin_prefetch(bsrc + (size_t)crow0 * I + (g + 1) * GS + ccol, 0, 1);
        }
    };
    auto wait_stage = [&]() {
        asm volatile("s_wait_asynccnt 0" ::: "memory");
        __syncthreads();
    };

    v8f facc[2][4];
    const v8f fzero = {};
    const v8i izero = {};
#pragma unroll
    for (int mi = 0; mi < 2; ++mi)
#pragma unroll
        for (int ni = 0; ni < 4; ++ni) facc[mi][ni] = fzero;

    issue_stage(0, 0);
    wait_stage();

    for (int g = 0; g < NG; ++g) {
        const int buf = g & 1;
        if (g + 1 < NG) issue_stage(g + 1, (g + 1) & 1);

        const char* atile = lds + buf * ATILE_BYTES;
        const char* btile = lds + BBASE + buf * ATILE_BYTES;

        v8i iacc[2][4];
#pragma unroll
        for (int mi = 0; mi < 2; ++mi)
#pragma unroll
            for (int ni = 0; ni < 4; ++ni) iacc[mi][ni] = izero;

#pragma unroll
        for (int kk = 0; kk < 4; ++kk) {
            v8i a[2], b[4];
#pragma unroll
            for (int mi = 0; mi < 2; ++mi) {
                const uint2* p = (const uint2*)(atile
                    + (mw * 32 + mi * 16 + ln) * LDA + hi * 8 + kk * 64);
                uint2 d0 = p[0], d1 = p[2], d2 = p[4], d3 = p[6];
                a[mi][0] = (int)d0.x; a[mi][1] = (int)d0.y;
                a[mi][2] = (int)d1.x; a[mi][3] = (int)d1.y;
                a[mi][4] = (int)d2.x; a[mi][5] = (int)d2.y;
                a[mi][6] = (int)d3.x; a[mi][7] = (int)d3.y;
            }
#pragma unroll
            for (int ni = 0; ni < 4; ++ni) {
                const uint2* p = (const uint2*)(btile
                    + (nw * 64 + ni * 16 + ln) * LDA + hi * 8 + kk * 64);
                uint2 d0 = p[0], d1 = p[2], d2 = p[4], d3 = p[6];
                b[ni][0] = (int)d0.x; b[ni][1] = (int)d0.y;
                b[ni][2] = (int)d1.x; b[ni][3] = (int)d1.y;
                b[ni][4] = (int)d2.x; b[ni][5] = (int)d2.y;
                b[ni][6] = (int)d3.x; b[ni][7] = (int)d3.y;
            }
#pragma unroll
            for (int mi = 0; mi < 2; ++mi)
#pragma unroll
                for (int ni = 0; ni < 4; ++ni)
                    iacc[mi][ni] = __builtin_amdgcn_wmma_i32_16x16x64_iu8(
                        true, a[mi], true, b[ni], iacc[mi][ni], false, false);
        }

        float rsf[2][8];
#pragma unroll
        for (int mi = 0; mi < 2; ++mi)
#pragma unroll
            for (int j = 0; j < 8; ++j)
                rsf[mi][j] = (float)rsx[(mbase + mi * 16 + hi * 8 + j) * NG + g];
#pragma unroll
        for (int ni = 0; ni < 4; ++ni) {
            float2 wm = wmeta[orow[ni] * NG + g];
#pragma unroll
            for (int mi = 0; mi < 2; ++mi)
#pragma unroll
                for (int j = 0; j < 8; ++j)
                    facc[mi][ni][j] += wm.x * (float)iacc[mi][ni][j]
                                     - wm.y * rsf[mi][j];
        }

        if (g + 1 < NG) wait_stage();
    }

    float w0n[4];
#pragma unroll
    for (int ni = 0; ni < 4; ++ni) w0n[ni] = W0[orow[ni]];
#pragma unroll
    for (int mi = 0; mi < 2; ++mi)
#pragma unroll
        for (int j = 0; j < 8; ++j) {
            int tt = mbase + mi * 16 + hi * 8 + j;
            float zp_ = zpx[tt];
            float sx_ = sx[tt];
            size_t row = (size_t)tt * O;
#pragma unroll
            for (int ni = 0; ni < 4; ++ni)
                out[row + orow[ni]] = sx_ * (facc[mi][ni][j] - zp_ * w0n[ni]);
        }
}

extern "C" void kernel_launch(void* const* d_in, const int* in_sizes, int n_in,
                              void* d_out, int out_size, void* d_ws, size_t ws_size,
                              hipStream_t stream) {
    (void)n_in; (void)out_size; (void)ws_size;
    const float* x      = (const float*)d_in[0];
    const int*   w      = (const int*)d_in[1];
    const float* scales = (const float*)d_in[2];
    const float* zeros  = (const float*)d_in[3];
    float* out = (float*)d_out;

    const int I = TOK_I;
    const int T = in_sizes[0] / I;
    const int O = in_sizes[1] / I;

    char* ws = (char*)d_ws;
    size_t off = 0;
    auto alloc = [&](size_t bytes) -> void* {
        void* p = ws + off;
        off += (bytes + 255) & ~(size_t)255;
        return p;
    };
    signed char* qx    = (signed char*)alloc((size_t)T * I);
    signed char* qw    = (signed char*)alloc((size_t)O * I);
    float*       sx    = (float*)alloc((size_t)T * sizeof(float));
    float*       zpx   = (float*)alloc((size_t)T * sizeof(float));
    int*         rsx   = (int*)alloc((size_t)T * NG * sizeof(int));
    float2*      wmeta = (float2*)alloc((size_t)O * NG * sizeof(float2));
    float*       W0    = (float*)alloc((size_t)O * sizeof(float));

    quant_x_kernel<<<T, 256, 0, stream>>>(x, qx, sx, zpx, rsx);
    prep_w_kernel<<<O, 256, 0, stream>>>(w, scales, zeros, qw, wmeta, W0);

    dim3 grid(T / 128, O / 128);
    gemm_kernel<<<grid, 256, LDS_TOTAL, stream>>>(qx, qw, sx, zpx, rsx, wmeta,
                                                  W0, out, O);
}
